// MambaBackboneComponent_90065464197198
// MI455X (gfx1250) — compile-verified, matrix operand order checked by reading, not run on hardware
//
#include <hip/hip_runtime.h>
#include <hip/hip_bf16.h>

// ---------------------------------------------------------------------------
// Problem constants (from the reference)
// ---------------------------------------------------------------------------
#define DMODEL   1024
#define DINNER   2048
#define DSTATE   16
#define DCONV    4
#define DTRANK   64
#define NLAYER   8
#define INDIM    256
#define BBATCH   16
#define TTT      64
#define SIN      13
#define SSEQ     16              // SIN + 3 tokens
#define LSEQ     (TTT * SSEQ)    // 1024 per batch
#define NROWS    (BBATCH * LSEQ) // 16384 activation rows
#define HROWS    (BBATCH * TTT * SIN) // 13312 embed rows
#define XPN      96              // x_proj true rows
#define XPNP     128             // x_proj padded rows (GEMM N, proj stride)

typedef unsigned short u16;
typedef __attribute__((ext_vector_type(16))) __bf16 v16bf;
typedef __attribute__((ext_vector_type(8)))  float  v8f;

struct __align__(16) U4 { unsigned int a, b, c, d; };
struct B32 { U4 lo, hi; };   // 32 bytes = one 16xbf16 WMMA fragment

static __device__ __forceinline__ u16 f2bf(float f) {
  unsigned int u = __float_as_uint(f);
  unsigned int r = (u + 0x7FFFu + ((u >> 16) & 1u)) >> 16;   // RNE
  return (u16)r;
}

static __device__ __forceinline__ v16bf frag_from(const u16* p) {
  B32 t;
  t.lo = *reinterpret_cast<const U4*>(p);
  t.hi = *reinterpret_cast<const U4*>(p + 16);
  return __builtin_bit_cast(v16bf, t);
}

// ---------------------------------------------------------------------------
// f32 -> bf16 (grid-stride)
// ---------------------------------------------------------------------------
__global__ void k_f32_to_bf16(const float* __restrict__ s,
                              u16* __restrict__ d, size_t n) {
  size_t i = (size_t)blockIdx.x * blockDim.x + threadIdx.x;
  size_t st = (size_t)gridDim.x * blockDim.x;
  for (; i < n; i += st) d[i] = f2bf(s[i]);
}

// x_proj (NLAYER, 96, 2048) f32 -> (NLAYER, 128, 2048) bf16, rows 96..127 = 0
__global__ void k_cvt_xproj(const float* __restrict__ s,
                            u16* __restrict__ d) {
  size_t i = (size_t)blockIdx.x * blockDim.x + threadIdx.x;
  size_t st = (size_t)gridDim.x * blockDim.x;
  const size_t total = (size_t)NLAYER * XPNP * DINNER;
  for (; i < total; i += st) {
    size_t l = i / ((size_t)XPNP * DINNER);
    size_t rem = i - l * ((size_t)XPNP * DINNER);
    size_t n = rem / DINNER;
    size_t k = rem - n * DINNER;
    d[i] = (n < XPN) ? f2bf(s[(l * XPN + n) * DINNER + k]) : (u16)0;
  }
}

// ---------------------------------------------------------------------------
// bf16 GEMM: C[M,N] (f32) = A[M,K] * B[N,K]^T, bf16 inputs. K compile-time.
// REQUIRES: M % 128 == 0, N % 128 == 0, K % 64 == 0 (caller guarantees).
// Block = 256 threads = 8 waves; block tile 128x128; wave tile 64x32 (4x2).
// 2-deep register pipeline with a branch-free loop body: each iteration is
// {MMA set0; load set0@kb+64; MMA set1; load set1@kb+96} in ONE basic block,
// so accumulator phis coalesce with WMMA D=C (no v_mov shuffling) and up to
// 24 global_load_b128 stay in flight over the matrix pipe.
// WMMA 16x16x32 bf16 fragment per lane: row = lane&15, K chunks at
// hk..hk+7 and hk+16..hk+23 with hk = (lane>>4)*8 (two 16B loads, K-contig).
// ---------------------------------------------------------------------------
template <int K>
__global__ __launch_bounds__(256) void k_gemm_bf16(
    const u16* __restrict__ A, const u16* __restrict__ B,
    float* __restrict__ C, int M, int N) {
  static_assert(K % 64 == 0, "K must be a multiple of 64");
  const int lane = threadIdx.x & 31;
  const int wave = threadIdx.x >> 5;
  const int wm = wave >> 2;             // 0..1
  const int wn = wave & 3;              // 0..3
  const int fr = lane & 15;             // row/col inside 16x16 tile
  const int hk = (lane >> 4) << 3;      // K sub-offset: 0 or 8

  const int row0 = blockIdx.x * 128 + wm * 64;
  const int col0 = blockIdx.y * 128 + wn * 32;

  v8f acc[4][2];
#pragma unroll
  for (int i = 0; i < 4; ++i)
#pragma unroll
    for (int j = 0; j < 2; ++j) { v8f z = {}; acc[i][j] = z; }

  // Loop-invariant 32-bit element offsets (byte offsets < 2^32 for all shapes)
  unsigned offA[4], offB[2];
#pragma unroll
  for (int i = 0; i < 4; ++i)
    offA[i] = (unsigned)(row0 + i * 16 + fr) * (unsigned)K + (unsigned)hk;
#pragma unroll
  for (int j = 0; j < 2; ++j)
    offB[j] = (unsigned)(col0 + j * 16 + fr) * (unsigned)K + (unsigned)hk;

  v16bf a0[4], b0[2], a1[4], b1[2];

#define LOAD_SET(AF, BF, KB)                                 \
  do {                                                       \
    _Pragma("unroll") for (int i = 0; i < 4; ++i)            \
      AF[i] = frag_from(A + offA[i] + (unsigned)(KB));       \
    _Pragma("unroll") for (int j = 0; j < 2; ++j)            \
      BF[j] = frag_from(B + offB[j] + (unsigned)(KB));       \
  } while (0)

#define MMA_SET(AF, BF)                                      \
  do {                                                       \
    _Pragma("unroll") for (int i = 0; i < 4; ++i)            \
      _Pragma("unroll") for (int j = 0; j < 2; ++j)          \
        acc[i][j] = __builtin_amdgcn_wmma_f32_16x16x32_bf16( \
            false, AF[i], false, BF[j], (short)0, acc[i][j], false, false); \
  } while (0)

  LOAD_SET(a0, b0, 0);
  LOAD_SET(a1, b1, 32);
#pragma unroll 1
  for (int kb = 0; kb + 64 < K; kb += 64) {
    MMA_SET(a0, b0);
    LOAD_SET(a0, b0, kb + 64);
    MMA_SET(a1, b1);
    LOAD_SET(a1, b1, kb + 96);
  }
  MMA_SET(a0, b0);
  MMA_SET(a1, b1);
#undef LOAD_SET
#undef MMA_SET

  // C/D layout: lane holds column n = lane&15, rows m0..m0+7 (m0=(lane>>4)*8)
  const int n = fr;
  const int m0 = (lane >> 4) << 3;
#pragma unroll
  for (int i = 0; i < 4; ++i)
#pragma unroll
    for (int j = 0; j < 2; ++j) {
      float* cp = C + (size_t)(row0 + i * 16 + m0) * N + (col0 + j * 16 + n);
#pragma unroll
      for (int r = 0; r < 8; ++r) cp[(size_t)r * N] = acc[i][j][r];
    }
}

// ---------------------------------------------------------------------------
// Token assembly: seq = concat(cls, h(+b_in), 0, 0) + positional encoding.
// hs := seq ; res := 0. One thread per (row, d) element of (16384, 1024).
// ---------------------------------------------------------------------------
__global__ __launch_bounds__(256) void k_assemble(
    const float* __restrict__ h, const float* __restrict__ cls,
    const float* __restrict__ b_in, float* __restrict__ hs,
    float* __restrict__ res) {
  size_t idx = (size_t)blockIdx.x * blockDim.x + threadIdx.x;
  int dd = (int)(idx & (DMODEL - 1));
  size_t row = idx >> 10;
  int b = (int)(row >> 10);
  int l = (int)(row & (LSEQ - 1));
  int t = l >> 4;
  int st = l & (SSEQ - 1);

  float val;
  if (st == 0) {
    val = cls[dd];
  } else if (st <= SIN) {
    size_t hrow = (size_t)((b * TTT + t) * SIN + (st - 1));
    val = h[hrow * DMODEL + dd] + b_in[dd];
  } else {
    val = 0.f;
  }
  int j2 = dd & ~1;
  float div = __expf((float)j2 * (-9.210340371976184f / (float)DMODEL));
  float ang = (float)t * div;
  float pe = (dd & 1) ? __cosf(ang) : __sinf(ang);
  hs[idx] = val + pe;
  res[idx] = 0.f;
}

// ---------------------------------------------------------------------------
// res += hs; hn = rmsnorm(res)*nw  -> bf16 GEMM input. One block per row.
// ---------------------------------------------------------------------------
__global__ __launch_bounds__(256) void k_res_rmsnorm(
    const float* __restrict__ hs, float* __restrict__ res,
    const float* __restrict__ nw, u16* __restrict__ hn_bf) {
  const int row = blockIdx.x;
  const int tid = threadIdx.x;
  __shared__ float red[256];
  size_t base = (size_t)row * DMODEL;
  float v[4];
  float ss = 0.f;
#pragma unroll
  for (int i = 0; i < 4; ++i) {
    int d = tid + i * 256;
    float r = hs[base + d] + res[base + d];
    res[base + d] = r;
    v[i] = r;
    ss += r * r;
  }
  red[tid] = ss;
  __syncthreads();
  for (int s = 128; s > 0; s >>= 1) {
    if (tid < s) red[tid] += red[tid + s];
    __syncthreads();
  }
  float scale = rsqrtf(red[0] * (1.f / (float)DMODEL) + 1e-5f);
#pragma unroll
  for (int i = 0; i < 4; ++i) {
    int d = tid + i * 256;
    hn_bf[base + d] = f2bf(v[i] * scale * nw[d]);
  }
}

// ---------------------------------------------------------------------------
// Depthwise causal conv (K=4) along L within each batch + bias + SiLU.
// xm = xz[:, :DINNER]. Emits f32 (scan input) and bf16 (x_proj GEMM input).
// ---------------------------------------------------------------------------
__global__ __launch_bounds__(256) void k_conv_silu(
    const float* __restrict__ xz, const float* __restrict__ cw,
    const float* __restrict__ cb, float* __restrict__ xc,
    u16* __restrict__ xc_bf) {
  size_t idx = (size_t)blockIdx.x * blockDim.x + threadIdx.x;  // NROWS*DINNER
  int d = (int)(idx & (DINNER - 1));
  size_t row = idx >> 11;
  int b = (int)(row >> 10);
  int l = (int)(row & (LSEQ - 1));
  float acc = cb[d];
#pragma unroll
  for (int k = 0; k < DCONV; ++k) {
    int ll = l - (DCONV - 1) + k;
    if (ll >= 0)
      acc += cw[d * DCONV + k] *
             xz[(((size_t)(b << 10) + ll) << 12) + d];  // xz row stride 4096
  }
  float s = acc / (1.f + __expf(-acc));
  xc[idx] = s;
  xc_bf[idx] = f2bf(s);
}

// ---------------------------------------------------------------------------
// proj[:, :64] -> bf16 (dt GEMM input); proj has row stride XPNP=128
// ---------------------------------------------------------------------------
__global__ __launch_bounds__(256) void k_proj_slice(
    const float* __restrict__ proj, u16* __restrict__ p64) {
  size_t idx = (size_t)blockIdx.x * blockDim.x + threadIdx.x;  // NROWS*64
  int c = (int)(idx & (DTRANK - 1));
  size_t row = idx >> 6;
  p64[idx] = f2bf(proj[row * XPNP + c]);
}

// ---------------------------------------------------------------------------
// dt = softplus(dt + dtb) in place
// ---------------------------------------------------------------------------
__global__ __launch_bounds__(256) void k_softplus(
    float* __restrict__ dt, const float* __restrict__ dtb) {
  size_t idx = (size_t)blockIdx.x * blockDim.x + threadIdx.x;  // NROWS*DINNER
  int d = (int)(idx & (DINNER - 1));
  float x = dt[idx] + dtb[d];
  dt[idx] = (x > 20.f) ? x : log1pf(__expf(x));
}

// ---------------------------------------------------------------------------
// Selective scan, fused with D-skip and SiLU(z) gate. One thread per (b,d)
// channel; 16-wide state in VGPRs; B/C staged through LDS in 128-step chunks.
// Writes bf16 y for the out_proj GEMM. proj row stride = XPNP.
// ---------------------------------------------------------------------------
#define SCHUNK 128
__global__ __launch_bounds__(256) void k_scan(
    const float* __restrict__ dt, const float* __restrict__ xc,
    const float* __restrict__ proj, const float* __restrict__ xz,
    const float* __restrict__ Alog, const float* __restrict__ Dp,
    u16* __restrict__ y_bf) {
  const int b = blockIdx.x;                       // 0..15
  const int d = blockIdx.y * 256 + threadIdx.x;   // 0..2047
  const int tid = threadIdx.x;

  __shared__ float sB[SCHUNK][DSTATE];
  __shared__ float sC[SCHUNK][DSTATE];

  float A2[DSTATE], h[DSTATE];
#pragma unroll
  for (int s = 0; s < DSTATE; ++s) {
    A2[s] = -__expf(Alog[(size_t)d * DSTATE + s]) * 1.4426950408889634f;
    h[s] = 0.f;
  }
  const float dskip = Dp[d];

  for (int lc = 0; lc < LSEQ; lc += SCHUNK) {
    for (int t = tid; t < SCHUNK * 32; t += 256) {
      int step = t >> 5;
      int s5 = t & 31;
      float v = proj[((size_t)(b * LSEQ + lc + step)) * XPNP + DTRANK + s5];
      if (s5 < DSTATE) sB[step][s5] = v;
      else             sC[step][s5 - DSTATE] = v;
    }
    __syncthreads();

    for (int i = 0; i < SCHUNK; ++i) {
      size_t row = (size_t)b * LSEQ + lc + i;
      float dtv = dt[row * DINNER + d];
      float u   = xc[row * DINNER + d];
      float du  = dtv * u;
      float y = 0.f;
#pragma unroll
      for (int s = 0; s < DSTATE; ++s) {
        float dA = exp2f(dtv * A2[s]);
        h[s] = dA * h[s] + du * sB[i][s];
        y += h[s] * sC[i][s];
      }
      float zv = xz[(row << 12) + DINNER + d];
      float sil = zv / (1.f + __expf(-zv));
      y_bf[row * DINNER + d] = f2bf((y + dskip * u) * sil);
    }
    __syncthreads();
  }
}

// ---------------------------------------------------------------------------
// Final: rmsnorm(hs+res) on CLS rows only -> d_out (1024, 1024) f32
// ---------------------------------------------------------------------------
__global__ __launch_bounds__(256) void k_final_pool(
    const float* __restrict__ hs, const float* __restrict__ res,
    const float* __restrict__ nfw, float* __restrict__ out) {
  const int bt = blockIdx.x;   // 0..1023
  const int tid = threadIdx.x;
  __shared__ float red[256];
  int b = bt >> 6, t = bt & 63;
  size_t src = ((size_t)b * LSEQ + (size_t)t * SSEQ) * DMODEL;
  float v[4];
  float ss = 0.f;
#pragma unroll
  for (int i = 0; i < 4; ++i) {
    int d = tid + i * 256;
    float r = hs[src + d] + res[src + d];
    v[i] = r;
    ss += r * r;
  }
  red[tid] = ss;
  __syncthreads();
  for (int s = 128; s > 0; s >>= 1) {
    if (tid < s) red[tid] += red[tid + s];
    __syncthreads();
  }
  float scale = rsqrtf(red[0] * (1.f / (float)DMODEL) + 1e-5f);
#pragma unroll
  for (int i = 0; i < 4; ++i) {
    int d = tid + i * 256;
    out[(size_t)bt * DMODEL + d] = v[i] * scale * nfw[d];
  }
}

// ---------------------------------------------------------------------------
// Workspace layout (bytes)
// ---------------------------------------------------------------------------
#define OFF_INPROJ_BF  ((size_t)0)
#define OFF_XPROJ_BF   (OFF_INPROJ_BF + (size_t)NLAYER*4096*1024*2)
#define OFF_DTPROJ_BF  (OFF_XPROJ_BF  + (size_t)NLAYER*XPNP*DINNER*2)
#define OFF_OUTPROJ_BF (OFF_DTPROJ_BF + (size_t)NLAYER*2048*64*2)
#define OFF_WIN_BF     (OFF_OUTPROJ_BF+ (size_t)NLAYER*1024*2048*2)
#define OFF_X_BF       (OFF_WIN_BF    + (size_t)1024*256*2)
#define OFF_HS         (OFF_X_BF      + (size_t)HROWS*256*2)
#define OFF_RES        (OFF_HS        + (size_t)NROWS*DMODEL*4)
#define OFF_HN_BF      (OFF_RES       + (size_t)NROWS*DMODEL*4)
#define OFF_XZ         (OFF_HN_BF     + (size_t)NROWS*DMODEL*2)
#define OFF_HEMB       OFF_XZ   /* h_embed overlays xz (disjoint lifetime) */
#define OFF_XC         (OFF_XZ        + (size_t)NROWS*2*DINNER*4)
#define OFF_XC_BF      (OFF_XC        + (size_t)NROWS*DINNER*4)
#define OFF_PROJ       (OFF_XC_BF     + (size_t)NROWS*DINNER*2)
#define OFF_P64_BF     (OFF_PROJ      + (size_t)NROWS*XPNP*4)
#define OFF_DT         (OFF_P64_BF    + (size_t)NROWS*DTRANK*2)
#define OFF_Y_BF       (OFF_DT        + (size_t)NROWS*DINNER*4)

extern "C" void kernel_launch(void* const* d_in, const int* in_sizes, int n_in,
                              void* d_out, int out_size, void* d_ws,
                              size_t ws_size, hipStream_t stream) {
  const float* x         = (const float*)d_in[0];
  // d_in[1] = tt (compile-time constant here)
  const float* w_in      = (const float*)d_in[2];
  const float* b_in      = (const float*)d_in[3];
  const float* cls       = (const float*)d_in[4];
  const float* norm_w    = (const float*)d_in[5];
  const float* in_proj_w = (const float*)d_in[6];
  const float* conv_w    = (const float*)d_in[7];
  const float* conv_b    = (const float*)d_in[8];
  const float* x_proj_w  = (const float*)d_in[9];
  const float* dt_proj_w = (const float*)d_in[10];
  const float* dt_proj_b = (const float*)d_in[11];
  const float* A_log     = (const float*)d_in[12];
  const float* D_param   = (const float*)d_in[13];
  const float* out_proj_w= (const float*)d_in[14];
  const float* normf_w   = (const float*)d_in[15];

  char* ws = (char*)d_ws;
  u16*   inproj_bf  = (u16*)(ws + OFF_INPROJ_BF);
  u16*   xproj_bf   = (u16*)(ws + OFF_XPROJ_BF);
  u16*   dtproj_bf  = (u16*)(ws + OFF_DTPROJ_BF);
  u16*   outproj_bf = (u16*)(ws + OFF_OUTPROJ_BF);
  u16*   win_bf     = (u16*)(ws + OFF_WIN_BF);
  u16*   x_bf       = (u16*)(ws + OFF_X_BF);
  float* hs         = (float*)(ws + OFF_HS);
  float* res        = (float*)(ws + OFF_RES);
  u16*   hn_bf      = (u16*)(ws + OFF_HN_BF);
  float* xz         = (float*)(ws + OFF_XZ);
  float* hemb       = (float*)(ws + OFF_HEMB);
  float* xc         = (float*)(ws + OFF_XC);
  u16*   xc_bf      = (u16*)(ws + OFF_XC_BF);
  float* proj       = (float*)(ws + OFF_PROJ);
  u16*   p64_bf     = (u16*)(ws + OFF_P64_BF);
  float* dtbuf      = (float*)(ws + OFF_DT);
  u16*   y_bf       = (u16*)(ws + OFF_Y_BF);

  const int CVT_BLK = 4096;
  // Weight/input conversions to bf16 (all fit L2 in bf16)
  k_f32_to_bf16<<<CVT_BLK, 256, 0, stream>>>(x, x_bf, (size_t)HROWS * 256);
  k_f32_to_bf16<<<CVT_BLK, 256, 0, stream>>>(w_in, win_bf, (size_t)1024 * 256);
  k_f32_to_bf16<<<CVT_BLK, 256, 0, stream>>>(in_proj_w, inproj_bf,
                                             (size_t)NLAYER * 4096 * 1024);
  k_cvt_xproj<<<CVT_BLK, 256, 0, stream>>>(x_proj_w, xproj_bf);
  k_f32_to_bf16<<<CVT_BLK, 256, 0, stream>>>(dt_proj_w, dtproj_bf,
                                             (size_t)NLAYER * 2048 * 64);
  k_f32_to_bf16<<<CVT_BLK, 256, 0, stream>>>(out_proj_w, outproj_bf,
                                             (size_t)NLAYER * 1024 * 2048);

  // Embed: h = x @ w_in^T   (13312 x 1024, K=256)
  k_gemm_bf16<256><<<dim3(HROWS / 128, DMODEL / 128), 256, 0, stream>>>(
      x_bf, win_bf, hemb, HROWS, DMODEL);

  // Tokens + positional encoding -> hs ; res = 0
  k_assemble<<<(NROWS * DMODEL) / 256, 256, 0, stream>>>(hemb, cls, b_in, hs,
                                                         res);

  for (int l = 0; l < NLAYER; ++l) {
    k_res_rmsnorm<<<NROWS, 256, 0, stream>>>(hs, res, norm_w + l * DMODEL,
                                             hn_bf);
    // xz = hn @ in_proj^T  (16384 x 4096, K=1024)
    k_gemm_bf16<DMODEL>
        <<<dim3(NROWS / 128, (2 * DINNER) / 128), 256, 0, stream>>>(
            hn_bf, inproj_bf + (size_t)l * 4096 * 1024, xz, NROWS, 2 * DINNER);
    k_conv_silu<<<(NROWS * DINNER) / 256, 256, 0, stream>>>(
        xz, conv_w + (size_t)l * DINNER * DCONV, conv_b + (size_t)l * DINNER,
        xc, xc_bf);
    // proj = xc @ x_proj_pad^T  (16384 x 128, K=2048), rows 96..127 are zero
    k_gemm_bf16<DINNER><<<dim3(NROWS / 128, XPNP / 128), 256, 0, stream>>>(
        xc_bf, xproj_bf + (size_t)l * XPNP * DINNER, proj, NROWS, XPNP);
    k_proj_slice<<<(NROWS * DTRANK) / 256, 256, 0, stream>>>(proj, p64_bf);
    // dt = proj[:, :64] @ dt_proj^T  (16384 x 2048, K=64)
    k_gemm_bf16<DTRANK><<<dim3(NROWS / 128, DINNER / 128), 256, 0, stream>>>(
        p64_bf, dtproj_bf + (size_t)l * 2048 * 64, dtbuf, NROWS, DINNER);
    k_softplus<<<(NROWS * DINNER) / 256, 256, 0, stream>>>(
        dtbuf, dt_proj_b + (size_t)l * DINNER);
    k_scan<<<dim3(BBATCH, DINNER / 256), 256, 0, stream>>>(
        dtbuf, xc, proj, xz, A_log + (size_t)l * DINNER * DSTATE,
        D_param + (size_t)l * DINNER, y_bf);
    // hs = y @ out_proj^T  (16384 x 1024, K=2048)
    k_gemm_bf16<DINNER><<<dim3(NROWS / 128, DMODEL / 128), 256, 0, stream>>>(
        y_bf, outproj_bf + (size_t)l * 1024 * 2048, hs, NROWS, DMODEL);
  }

  k_final_pool<<<BBATCH * TTT, 256, 0, stream>>>(hs, res, normf_w,
                                                 (float*)d_out);
}
